// LinkPredictorC_64493228917219
// MI455X (gfx1250) — matrix operand order repaired and checked by reading, not compiled, not run on hardware
//
#include <hip/hip_runtime.h>
#include <hip/hip_bf16.h>

typedef __attribute__((ext_vector_type(16))) _Float16 v16h;
typedef __attribute__((ext_vector_type(8)))  _Float16 v8h;
typedef __attribute__((ext_vector_type(8)))  float    v8f;

#define N_NODES_C 100000
#define IN_CH_C   128
#define HID_C     256
#define EMB_C     128
#define N_PAIRS_C 500000
#define PAIR_K_PAD 544   // 514 padded to 17*32

// LDS row pads chosen so dword stride % 64 == 4 (20 for LP_LD) -> conflict-free A-frag loads
#define LA_LD  136   // 128 + 8   (272 B/row)
#define LH_LD  264   // 256 + 8   (528 B/row)
#define LP_LD  552   // 544 + 8   (1104 B/row)
#define LS1_LD 264
#define LS2_LD 136

// ---------------------------------------------------------------------------
// WMMA helper (gfx1250, wave32). D = A(16x32 f16) x B(32x16 f16) + C(f32)
// ---------------------------------------------------------------------------
__device__ __forceinline__ v8f wmma_f16(v16h a, v16h b, v8f c) {
  return __builtin_amdgcn_wmma_f32_16x16x32_f16(
      false, a, false, b, (short)0, c, false, false);
}

// A fragment from a row-major f16 LDS tile [M][ldk] (ISA 16-bit A layout).
// Two aligned 16B LDS loads per lane.
__device__ __forceinline__ v16h load_A_frag(const _Float16* lds, int ldk,
                                            int mtile, int ktile, int lane) {
  int m  = (lane & 15) + mtile * 16;
  int kb = ktile * 32 + ((lane >> 4) & 1) * 8;
  const _Float16* row = lds + m * ldk + kb;
  v8h lo = *(const v8h*)(row);
  v8h hi = *(const v8h*)(row + 16);
  v16h a;
#pragma unroll
  for (int i = 0; i < 8; ++i) { a[i] = lo[i]; a[8 + i] = hi[i]; }
  return a;
}

// B fragment from pre-packed global weights: 32B contiguous per lane.
__device__ __forceinline__ v16h load_B_frag(const _Float16* packed,
                                            int frag, int lane) {
  return *(const v16h*)(packed + ((size_t)frag * 32 + lane) * 16);
}

// ---------------------------------------------------------------------------
// Weight packing: row-major f32 W[K][N] -> WMMA B-fragment-ordered f16.
// B layout (32x16): lane l holds column N=l&15; half e holds K=(e&7)+(l>=16)*8+(e>>3)*16.
// ---------------------------------------------------------------------------
__global__ void pack_b_kernel(const float* __restrict__ W, _Float16* __restrict__ out,
                              int K, int N, int Kpad) {
  int idx = blockIdx.x * blockDim.x + threadIdx.x;
  int total = (Kpad / 32) * (N / 16) * 512;
  if (idx >= total) return;
  int e    = idx & 15;
  int lane = (idx >> 4) & 31;
  int frag = idx >> 9;
  int ntiles = N / 16;
  int nt = frag % ntiles;
  int kt = frag / ntiles;
  int k = kt * 32 + (e & 7) + ((lane >> 4) & 1) * 8 + (e >> 3) * 16;
  int n = nt * 16 + (lane & 15);
  float w = (k < K) ? W[(size_t)k * N + n] : 0.0f;
  out[idx] = (_Float16)w;
}

// ---------------------------------------------------------------------------
// Kernel 1: node MLP. 64 nodes / 256-thread block (8 waves, wave32).
// ---------------------------------------------------------------------------
__global__ __launch_bounds__(256) void node_mlp_kernel(
    const float* __restrict__ x, const float* __restrict__ x_mean,
    const float* __restrict__ x_std,
    const float* __restrict__ b1, const float* __restrict__ bn_g,
    const float* __restrict__ bn_b, const float* __restrict__ bn_mu,
    const float* __restrict__ bn_var, const float* __restrict__ b2,
    const _Float16* __restrict__ W1p, const _Float16* __restrict__ W2p,
    _Float16* __restrict__ z, int n_nodes) {
  __shared__ __align__(32) _Float16 lA[64][LA_LD];   // ~17 KB
  __shared__ __align__(32) _Float16 lH[64][LH_LD];   // ~33 KB
  __shared__ float sMean[IN_CH_C], sInv[IN_CH_C];
  const int tid = threadIdx.x, lane = tid & 31, wave = tid >> 5;
  const int row0 = blockIdx.x * 64;

  if (tid < IN_CH_C) {
    sMean[tid] = x_mean[tid];
    sInv[tid]  = 1.0f / x_std[tid];
  }
  __syncthreads();

  // Stage + normalize x tile (float4 loads)
  for (int i = tid; i < 64 * 32; i += 256) {
    int r = i >> 5, c4 = (i & 31) * 4;
    int node = row0 + r;
    float4 v = make_float4(0.f, 0.f, 0.f, 0.f);
    if (node < n_nodes) v = *(const float4*)(x + (size_t)node * IN_CH_C + c4);
    float vv[4] = {v.x, v.y, v.z, v.w};
#pragma unroll
    for (int j = 0; j < 4; ++j) {
      float t = vv[j];
      if (!isfinite(t)) t = 0.0f;
      t = (t - sMean[c4 + j]) * sInv[c4 + j];
      lA[r][c4 + j] = (_Float16)fminf(10.0f, fmaxf(-10.0f, t));
    }
  }
  __syncthreads();

  // GEMM1: M=64, N=256, K=128. wave -> mt = w>>1, 8 n-tiles each.
  {
    int mt  = wave >> 1;
    int ntb = (wave & 1) * 8;
    v8f acc[8] = {};
#pragma unroll
    for (int kt = 0; kt < 4; ++kt) {
      v16h a = load_A_frag(&lA[0][0], LA_LD, mt, kt, lane);
#pragma unroll
      for (int j = 0; j < 8; ++j) {
        v16h b = load_B_frag(W1p, kt * 16 + ntb + j, lane);
        acc[j] = wmma_f16(a, b, acc[j]);
      }
    }
    int mbase = mt * 16 + ((lane >> 4) & 1) * 8;
#pragma unroll
    for (int j = 0; j < 8; ++j) {
      int n = (ntb + j) * 16 + (lane & 15);
      float s  = bn_g[n] * rsqrtf(bn_var[n] + 1e-5f);
      float sh = (b1[n] - bn_mu[n]) * s + bn_b[n];
#pragma unroll
      for (int r = 0; r < 8; ++r)
        lH[mbase + r][n] = (_Float16)fmaxf(acc[j][r] * s + sh, 0.0f);
    }
  }
  __syncthreads();

  // GEMM2: M=64, N=128, K=256. wave -> mt = w>>1, 4 n-tiles each.
  {
    int mt  = wave >> 1;
    int ntb = (wave & 1) * 4;
    v8f acc[4] = {};
    for (int kt = 0; kt < 8; ++kt) {
      v16h a = load_A_frag(&lH[0][0], LH_LD, mt, kt, lane);
#pragma unroll
      for (int j = 0; j < 4; ++j) {
        v16h b = load_B_frag(W2p, kt * 8 + ntb + j, lane);
        acc[j] = wmma_f16(a, b, acc[j]);
      }
    }
    int mbase = mt * 16 + ((lane >> 4) & 1) * 8;
#pragma unroll
    for (int j = 0; j < 4; ++j) {
      int n = (ntb + j) * 16 + (lane & 15);
      float bb = b2[n];
#pragma unroll
      for (int r = 0; r < 8; ++r) {
        int node = row0 + mbase + r;
        if (node < n_nodes)
          z[(size_t)node * EMB_C + n] = (_Float16)fmaxf(acc[j][r] + bb, 0.0f);
      }
    }
  }
}

// ---------------------------------------------------------------------------
// Kernel 2: pair MLP. 64 pairs / 256-thread block.
// ---------------------------------------------------------------------------
__global__ __launch_bounds__(256) void pair_mlp_kernel(
    const int* __restrict__ pairs, const float* __restrict__ logdeg,
    const _Float16* __restrict__ z,
    const float* __restrict__ Sb1, const float* __restrict__ Sb2,
    const float* __restrict__ SW3, const float* __restrict__ Sb3,
    const _Float16* __restrict__ SW1p, const _Float16* __restrict__ SW2p,
    float* __restrict__ out, int n_pairs) {
  __shared__ __align__(32) _Float16 lP[64][LP_LD];    // ~69 KB
  __shared__ __align__(32) _Float16 lS1[64][LS1_LD];  // ~33 KB
  __shared__ __align__(32) _Float16 lS2[64][LS2_LD];  // ~17 KB
  const int tid = threadIdx.x, lane = tid & 31, wave = tid >> 5;
  const int p0 = blockIdx.x * 64;

  // Build pair features: 4 threads per pair row, 32 cols each, v8h loads
  {
    int r  = tid >> 2;
    int cq = (tid & 3) * 32;
    int p  = p0 + r;
    int si = 0, di = 0;
    if (p < n_pairs) { si = pairs[2 * p]; di = pairs[2 * p + 1]; }
    const _Float16* zs = z + (size_t)si * EMB_C;
    const _Float16* zd = z + (size_t)di * EMB_C;
#pragma unroll
    for (int c = cq; c < cq + 32; c += 8) {
      v8h a8 = *(const v8h*)(zs + c);
      v8h b8 = *(const v8h*)(zd + c);
      v8h m8, d8;
#pragma unroll
      for (int i = 0; i < 8; ++i) {
        float a = (float)a8[i], b = (float)b8[i];
        m8[i] = (_Float16)(a * b);
        d8[i] = (_Float16)fabsf(a - b);
      }
      *(v8h*)&lP[r][c]       = a8;
      *(v8h*)&lP[r][128 + c] = b8;
      *(v8h*)&lP[r][256 + c] = m8;
      *(v8h*)&lP[r][384 + c] = d8;
    }
    if ((tid & 3) == 0) {
      lP[r][512] = (_Float16)((p < n_pairs) ? logdeg[si] : 0.0f);
      lP[r][513] = (_Float16)((p < n_pairs) ? logdeg[di] : 0.0f);
#pragma unroll
      for (int c = 514; c < 544; ++c) lP[r][c] = (_Float16)0.0f;
    }
  }
  __syncthreads();

  // GEMM1: M=64, N=256, K=544 (17 kt). wave -> mt = w>>1, 8 n-tiles, reg-blocked.
  {
    int mt  = wave >> 1;
    int ntb = (wave & 1) * 8;
    v8f acc[8] = {};
    for (int kt = 0; kt < 17; ++kt) {
      v16h a = load_A_frag(&lP[0][0], LP_LD, mt, kt, lane);
#pragma unroll
      for (int j = 0; j < 8; ++j) {
        v16h b = load_B_frag(SW1p, kt * 16 + ntb + j, lane);
        acc[j] = wmma_f16(a, b, acc[j]);
      }
    }
    int mbase = mt * 16 + ((lane >> 4) & 1) * 8;
#pragma unroll
    for (int j = 0; j < 8; ++j) {
      int n = (ntb + j) * 16 + (lane & 15);
      float bb = Sb1[n];
#pragma unroll
      for (int r = 0; r < 8; ++r)
        lS1[mbase + r][n] = (_Float16)fmaxf(acc[j][r] + bb, 0.0f);
    }
  }
  __syncthreads();

  // GEMM2: M=64, N=128, K=256 (8 kt). wave -> mt = w>>1, 4 n-tiles.
  {
    int mt  = wave >> 1;
    int ntb = (wave & 1) * 4;
    v8f acc[4] = {};
    for (int kt = 0; kt < 8; ++kt) {
      v16h a = load_A_frag(&lS1[0][0], LS1_LD, mt, kt, lane);
#pragma unroll
      for (int j = 0; j < 4; ++j) {
        v16h b = load_B_frag(SW2p, kt * 8 + ntb + j, lane);
        acc[j] = wmma_f16(a, b, acc[j]);
      }
    }
    int mbase = mt * 16 + ((lane >> 4) & 1) * 8;
#pragma unroll
    for (int j = 0; j < 4; ++j) {
      int n = (ntb + j) * 16 + (lane & 15);
      float bb = Sb2[n];
#pragma unroll
      for (int r = 0; r < 8; ++r)
        lS2[mbase + r][n] = (_Float16)fmaxf(acc[j][r] + bb, 0.0f);
    }
  }
  __syncthreads();

  // Final 128->1 projection + nan_to_num(0, +20, -20)
  if (tid < 64) {
    int p = p0 + tid;
    float acc = Sb3[0];
    for (int cb = 0; cb < EMB_C; cb += 8) {
      v8h v = *(const v8h*)&lS2[tid][cb];
#pragma unroll
      for (int i = 0; i < 8; ++i) acc += (float)v[i] * SW3[cb + i];
    }
    if (isnan(acc)) acc = 0.0f;
    else if (isinf(acc)) acc = (acc > 0.0f) ? 20.0f : -20.0f;
    if (p < n_pairs) out[p] = acc;
  }
}

// ---------------------------------------------------------------------------
// Launch: pack weights -> node MLP -> pair MLP (stream-ordered).
// Workspace layout (bytes):
//   z    f16 [100000][128] @ 0           (25,600,000)
//   W1p  f16 packed        @ 25,600,000  (   65,536)
//   W2p  f16 packed        @ 25,665,536  (   65,536)
//   SW1p f16 packed        @ 25,731,072  (  278,528)
//   SW2p f16 packed        @ 26,009,600  (   65,536)
// ---------------------------------------------------------------------------
extern "C" void kernel_launch(void* const* d_in, const int* in_sizes, int n_in,
                              void* d_out, int out_size, void* d_ws, size_t ws_size,
                              hipStream_t stream) {
  const float* x      = (const float*)d_in[0];
  // d_in[1] = edge_index (unused by reference)
  const int*   pairs  = (const int*)d_in[2];
  const float* x_mean = (const float*)d_in[3];
  const float* x_std  = (const float*)d_in[4];
  const float* logdeg = (const float*)d_in[5];
  const float* W1     = (const float*)d_in[6];
  const float* b1     = (const float*)d_in[7];
  const float* bn_g   = (const float*)d_in[8];
  const float* bn_b   = (const float*)d_in[9];
  const float* bn_mu  = (const float*)d_in[10];
  const float* bn_var = (const float*)d_in[11];
  const float* W2     = (const float*)d_in[12];
  const float* b2     = (const float*)d_in[13];
  const float* SW1    = (const float*)d_in[14];
  const float* Sb1    = (const float*)d_in[15];
  const float* SW2    = (const float*)d_in[16];
  const float* Sb2    = (const float*)d_in[17];
  const float* SW3    = (const float*)d_in[18];
  const float* Sb3    = (const float*)d_in[19];
  float* out = (float*)d_out;

  char* ws = (char*)d_ws;
  _Float16* zf16 = (_Float16*)(ws + 0);
  _Float16* W1p  = (_Float16*)(ws + 25600000);
  _Float16* W2p  = (_Float16*)(ws + 25665536);
  _Float16* SW1p = (_Float16*)(ws + 25731072);
  _Float16* SW2p = (_Float16*)(ws + 26009600);

  int tot;
  tot = (IN_CH_C / 32) * (HID_C / 16) * 512;
  pack_b_kernel<<<(tot + 255) / 256, 256, 0, stream>>>(W1, W1p, IN_CH_C, HID_C, IN_CH_C);
  tot = (HID_C / 32) * (EMB_C / 16) * 512;
  pack_b_kernel<<<(tot + 255) / 256, 256, 0, stream>>>(W2, W2p, HID_C, EMB_C, HID_C);
  tot = (PAIR_K_PAD / 32) * (HID_C / 16) * 512;
  pack_b_kernel<<<(tot + 255) / 256, 256, 0, stream>>>(SW1, SW1p, 514, HID_C, PAIR_K_PAD);
  tot = (HID_C / 32) * (EMB_C / 16) * 512;
  pack_b_kernel<<<(tot + 255) / 256, 256, 0, stream>>>(SW2, SW2p, HID_C, EMB_C, HID_C);

  node_mlp_kernel<<<(N_NODES_C + 63) / 64, 256, 0, stream>>>(
      x, x_mean, x_std, b1, bn_g, bn_b, bn_mu, bn_var, b2, W1p, W2p,
      zf16, N_NODES_C);

  pair_mlp_kernel<<<(N_PAIRS_C + 63) / 64, 256, 0, stream>>>(
      pairs, logdeg, zf16, Sb1, Sb2, SW3, Sb3, SW1p, SW2p, out, N_PAIRS_C);
}
